// Attention_28458453303729
// MI455X (gfx1250) — matrix operand order repaired and checked by reading, not compiled, not run on hardware
//
#include <hip/hip_runtime.h>

typedef __attribute__((ext_vector_type(16))) _Float16 v16h;
typedef __attribute__((ext_vector_type(8)))  _Float16 v8h;
typedef __attribute__((ext_vector_type(4)))  _Float16 v4h;
typedef __attribute__((ext_vector_type(8)))  float    v8f;
typedef __attribute__((ext_vector_type(4)))  unsigned int u32x4;
typedef __attribute__((ext_vector_type(8)))  int      i32x8;
typedef __attribute__((ext_vector_type(4)))  int      i32x4;

#define S_LEN 2048
#define NB    32
#define HDIM  1024
#define EDIM  1024
#define MTOT  (S_LEN * NB)     // 65536 GEMM rows
#define MTILE 128              // rows per block
#define LDSH  1032             // padded A row stride in halfs (516 dwords, 516%64==4)
#define KBLK  64               // K elements per TDM-staged B block
#define TBLK  (HDIM / KBLK)    // 16 K-blocks per pass
#define BROWB 144              // B row stride in bytes after TDM padding (128B + 16B pad)
#define BROWH (BROWB / 2)      // ... in halfs (72)
#define ABYTES (MTILE * LDSH * 2)        // 264192
#define BBYTES (128 * BROWB)             // 18432 per buffer
#define LDS_TOTAL (ABYTES + 2 * BBYTES)  // 301056 < 320K

// ---------------------------------------------------------------------------
// Prep 1: W2 (E,H) f32 -> f16
// ---------------------------------------------------------------------------
__global__ void cvt_w2_kernel(const float* __restrict__ W2, _Float16* __restrict__ W2h) {
    int i = (blockIdx.x * 256 + threadIdx.x) * 4;
    float4 f = *(const float4*)(W2 + i);
    v4h h;
    h[0] = (_Float16)f.x; h[1] = (_Float16)f.y;
    h[2] = (_Float16)f.z; h[3] = (_Float16)f.w;
    *(v4h*)(W2h + i) = h;
}

// ---------------------------------------------------------------------------
// Prep 2: t1[n][e] = sum_h hidden[n,h]*W1[e,h] + b1[e] + b2[e]   (tiny GEMM)
// ---------------------------------------------------------------------------
__global__ void t1_kernel(const float* __restrict__ hidden, const float* __restrict__ W1,
                          const float* __restrict__ b1, const float* __restrict__ b2,
                          float* __restrict__ t1) {
    int e = blockIdx.x * 256 + threadIdx.x;
    int n = blockIdx.y;
    const float* hr = hidden + n * HDIM;
    const float* wr = W1 + (size_t)e * HDIM;
    float acc = 0.f;
    for (int h = 0; h < HDIM; h += 4) {
        acc += hr[h]   * wr[h];
        acc += hr[h+1] * wr[h+1];
        acc += hr[h+2] * wr[h+2];
        acc += hr[h+3] * wr[h+3];
    }
    t1[n * EDIM + e] = acc + b1[e] + b2[e];
}

// ---------------------------------------------------------------------------
// TDM: TENSOR_LOAD_TO_LDS of a 2D B tile: 128 rows (e) x 64 halfs (k),
// row stride 1024 halfs in memory, padded to 144B rows in LDS (gcd(36,64)=4
// dword stride -> conflict-free 16-lane spread). Descriptor per ISA ch.8.
// ---------------------------------------------------------------------------
__device__ __forceinline__ void tdm_load_b(const _Float16* gsrc, unsigned ldsOff) {
    unsigned long long ga = (unsigned long long)(size_t)gsrc;
    u32x4 g0;
    g0[0] = 1u;                                   // count=1, is_restore=0, gather off
    g0[1] = ldsOff;                               // lds_addr (bytes)
    g0[2] = (unsigned)ga;                         // global_addr[31:0]
    g0[3] = (unsigned)(ga >> 32) | (2u << 30);    // global_addr[56:32] | type=2
    i32x8 g1;
    g1[0] = (1 << 16)      // data_size = 1 -> 2 bytes
          | (1 << 20)      // pad_enable
          | (4 << 22)      // pad_interval code 4 -> every 32 dwords (one 128B row)
          | (3 << 25);     // pad_amount   code 3 -> 4 dwords (16B)
    g1[1] = (HDIM & 0xffff) << 16;                // tensor_dim0 low16 (1024)
    g1[2] = (EDIM & 0xffff) << 16;                // tensor_dim0 hi=0 | tensor_dim1 low16
    g1[3] = (KBLK << 16);                         // tensor_dim1 hi=0 | tile_dim0=64
    g1[4] = 128;                                  // tile_dim1=128, tile_dim2=0
    g1[5] = HDIM;                                 // tensor_dim0_stride low32 = 1024
    g1[6] = 0;                                    // stride hi | tensor_dim1_stride lo
    g1[7] = 0;
    i32x4 z4 = {0, 0, 0, 0};
#if defined(__clang_major__) && (__clang_major__ >= 23)
    i32x8 z8 = {0, 0, 0, 0, 0, 0, 0, 0};
    __builtin_amdgcn_tensor_load_to_lds(g0, g1, z4, z4, z8, 0);
#else
    __builtin_amdgcn_tensor_load_to_lds(g0, g1, z4, z4, 0);
#endif
}

// ---------------------------------------------------------------------------
// Main fused kernel: a[gm] = b3 + sum_e tanh((A@W2h^T)[gm,e] + t1[gm%32,e])*W3[e]
// 8 wave32 per block. A tile (128x1024 f16) in LDS; B chunks DMA'd by the TDM
// into a double-buffered LDS region; batched ds_loads + 8 back-to-back WMMAs.
// ---------------------------------------------------------------------------
__global__ __launch_bounds__(256, 1)
void attn_gemm_kernel(const float* __restrict__ A, const _Float16* __restrict__ W2h,
                      const float* __restrict__ t1, const float* __restrict__ W3,
                      const float* __restrict__ b3, float* __restrict__ aOut) {
    extern __shared__ _Float16 smem[];            // [A: 258KB][B0: 18KB][B1: 18KB]
    _Float16* Ash = smem;
    _Float16* Bsh = smem + ABYTES / 2;            // halfs
    // LDS byte offset of B buffers: generic LDS addresses map via addr[31:0]
    const unsigned bOff0 = (unsigned)(size_t)(const void*)Bsh;

    const int tid  = threadIdx.x;
    const int wave = tid >> 5;
    const int lane = tid & 31;
    const int r    = lane & 15;
    const int hi   = lane >> 4;
    const int rowbase0 = blockIdx.x * MTILE;

    // ---- stage A tile (contiguous 512KB global), convert f32 -> f16 ----
    const float4* gA = (const float4*)(A + (size_t)rowbase0 * HDIM);
    for (int i = tid; i < (MTILE * HDIM) / 4; i += 256) {
        float4 f = gA[i];
        v4h h;
        h[0] = (_Float16)f.x; h[1] = (_Float16)f.y;
        h[2] = (_Float16)f.z; h[3] = (_Float16)f.w;
        int idx = i * 4;
        int row = idx >> 10, col = idx & (HDIM - 1);
        *(v4h*)(Ash + row * LDSH + col) = h;
    }
    __syncthreads();

    const int rowbase = rowbase0 + wave * 16;
    const _Float16* aRow = Ash + (wave * 16 + r) * LDSH;

    float rowacc[8];
    #pragma unroll
    for (int j = 0; j < 8; ++j) rowacc[j] = 0.f;

    for (int p = 0; p < 8; ++p) {                 // 8 passes x 128 columns of E
        v8f acc[8] = {};
        const _Float16* gB = W2h + (size_t)(p * 128) * HDIM;

        if (wave == 0) tdm_load_b(gB, bOff0);     // prologue: K-block 0 -> buf0

        for (int t = 0; t < TBLK; ++t) {
            if (wave == 0) {
                if (t + 1 < TBLK) {               // pipeline next K-block
                    tdm_load_b(gB + (t + 1) * KBLK, bOff0 + ((t + 1) & 1) * BBYTES);
                    __builtin_amdgcn_s_wait_tensorcnt((short)1);
                } else {
                    __builtin_amdgcn_s_wait_tensorcnt((short)0);
                }
            }
            __syncthreads();                      // buf[t&1] ready for all waves

            const _Float16* bb = Bsh + (t & 1) * (BBYTES / 2);
            const int kb = t * KBLK;
            #pragma unroll
            for (int kk2 = 0; kk2 < KBLK / 32; ++kk2) {
                const int k = kb + kk2 * 32;
                // A fragment (16-bit A 16x32 layout)
                v8h a0 = *(const v8h*)(aRow + k + hi * 8);
                v8h a1 = *(const v8h*)(aRow + k + hi * 8 + 16);
                v16h af;
                #pragma unroll
                for (int i = 0; i < 8; ++i) { af[i] = a0[i]; af[i + 8] = a1[i]; }

                // Batch-load ALL 8 B fragments (one ds clause, one wait), then
                // fire 8 independent WMMAs back-to-back.
                v16h bf[8];
                #pragma unroll
                for (int c = 0; c < 8; ++c) {
                    const _Float16* bp = bb + (c * 16 + r) * BROWH + kk2 * 32 + hi * 8;
                    v8h b0 = *(const v8h*)(bp);
                    v8h b1 = *(const v8h*)(bp + 16);
                    #pragma unroll
                    for (int i = 0; i < 8; ++i) { bf[c][i] = b0[i]; bf[c][i + 8] = b1[i]; }
                }
                #pragma unroll
                for (int c = 0; c < 8; ++c) {
                    acc[c] = __builtin_amdgcn_wmma_f32_16x16x32_f16(
                        false, af, false, bf[c], (short)0, acc[c], false, false);
                }
            }
            __syncthreads();                      // done reading buf[t&1]
        }

        // ---- fused epilogue: + t1, tanh, * W3, fold into per-row partials ----
        #pragma unroll
        for (int c = 0; c < 8; ++c) {
            const int e = (p * 8 + c) * 16 + r;
            const float w3v = W3[e];
            #pragma unroll
            for (int j = 0; j < 8; ++j) {
                const int m = hi * 8 + j;         // C layout: row = vgpr + 8*hi
                const int n = (rowbase + m) & (NB - 1);
                rowacc[j] += tanhf(acc[c][j] + t1[n * EDIM + e]) * w3v;
            }
        }
    }

    // ---- reduce E across the 16 column-lanes, write a[] ----
    const float b3v = b3[0];
    #pragma unroll
    for (int j = 0; j < 8; ++j) {
        float v = rowacc[j];
        v += __shfl_xor(v, 1);
        v += __shfl_xor(v, 2);
        v += __shfl_xor(v, 4);
        v += __shfl_xor(v, 8);
        if (r == 0) aOut[rowbase + hi * 8 + j] = v + b3v;
    }
}

// ---------------------------------------------------------------------------
// Softmax over S (one block per n); also zero-inits the c output region.
// ---------------------------------------------------------------------------
__global__ void softmax_kernel(const float* __restrict__ aIn, float* __restrict__ out) {
    __shared__ float red[256];
    const int n = blockIdx.x, t = threadIdx.x;
    float* cOut  = out;
    float* alpha = out + NB * HDIM;

    #pragma unroll
    for (int i = 0; i < HDIM / 256; ++i) cOut[n * HDIM + i * 256 + t] = 0.f;

    float vals[S_LEN / 256];
    float mx = -3.402823466e+38f;
    #pragma unroll
    for (int i = 0; i < S_LEN / 256; ++i) {
        float v = aIn[(i * 256 + t) * NB + n];
        vals[i] = v;
        mx = fmaxf(mx, v);
    }
    red[t] = mx; __syncthreads();
    for (int s2 = 128; s2 > 0; s2 >>= 1) {
        if (t < s2) red[t] = fmaxf(red[t], red[t + s2]);
        __syncthreads();
    }
    mx = red[0]; __syncthreads();

    float sum = 0.f;
    #pragma unroll
    for (int i = 0; i < S_LEN / 256; ++i) { vals[i] = expf(vals[i] - mx); sum += vals[i]; }
    red[t] = sum; __syncthreads();
    for (int s2 = 128; s2 > 0; s2 >>= 1) {
        if (t < s2) red[t] += red[t + s2];
        __syncthreads();
    }
    const float inv = 1.f / red[0];
    #pragma unroll
    for (int i = 0; i < S_LEN / 256; ++i)
        alpha[(i * 256 + t) * NB + n] = vals[i] * inv;
}

// ---------------------------------------------------------------------------
// c[n][h] = sum_s alpha[s,n] * out_e[s,n,h]   (split-S 16-way, f32 atomics)
// ---------------------------------------------------------------------------
__global__ void ctx_kernel(const float* __restrict__ out_e, const float* __restrict__ alpha,
                           float* __restrict__ cOut) {
    const int h  = blockIdx.x * 256 + threadIdx.x;
    const int n  = blockIdx.y;
    const int s0 = blockIdx.z * (S_LEN / 16);
    float acc = 0.f;
    for (int s = s0; s < s0 + S_LEN / 16; ++s) {
        const float al = alpha[s * NB + n];
        acc += al * out_e[((size_t)s * NB + n) * HDIM + h];
    }
    atomicAdd(&cOut[n * HDIM + h], acc);
}

// ---------------------------------------------------------------------------
extern "C" void kernel_launch(void* const* d_in, const int* in_sizes, int n_in,
                              void* d_out, int out_size, void* d_ws, size_t ws_size,
                              hipStream_t stream) {
    const float* out_e  = (const float*)d_in[0];
    const float* hidden = (const float*)d_in[1];
    const float* W1     = (const float*)d_in[2];
    const float* b1     = (const float*)d_in[3];
    const float* W2     = (const float*)d_in[4];
    const float* b2     = (const float*)d_in[5];
    const float* W3     = (const float*)d_in[6];
    const float* b3     = (const float*)d_in[7];
    float* out = (float*)d_out;

    char* ws = (char*)d_ws;
    _Float16* W2h = (_Float16*)ws;                                        // 2 MB
    float*    t1  = (float*)(ws + (size_t)EDIM * HDIM * 2);               // 128 KB
    float*    aB  = (float*)(ws + (size_t)EDIM * HDIM * 2 + NB * EDIM * 4); // 256 KB

    cvt_w2_kernel<<<(EDIM * HDIM / 4) / 256, 256, 0, stream>>>(W2, W2h);
    t1_kernel<<<dim3(EDIM / 256, NB), 256, 0, stream>>>(hidden, W1, b1, b2, t1);

    attn_gemm_kernel<<<MTOT / MTILE, 256, LDS_TOTAL, stream>>>(out_e, W2h, t1, W3, b3, aB);

    softmax_kernel<<<NB, 256, 0, stream>>>(aB, out);
    ctx_kernel<<<dim3(HDIM / 256, NB, 16), 256, 0, stream>>>(out_e, out + NB * HDIM, out);
}
